// VisualResolutionRouter_73581379715468
// MI455X (gfx1250) — matrix operand order repaired and checked by reading, not compiled, not run on hardware
//
#include <hip/hip_runtime.h>
#include <hip/hip_bf16.h>

typedef __attribute__((ext_vector_type(16))) __bf16 v16bf;
typedef __attribute__((ext_vector_type(8)))  __bf16 v8bf;
typedef __attribute__((ext_vector_type(8)))  float  v8f;

#define DD 1024
#define BB 8
#define LL 4096

__device__ __forceinline__ float clipf(float v, float lo, float hi) {
    return fminf(fmaxf(v, lo), hi);
}

// ---- WMMA helpers -----------------------------------------------------------
__device__ __forceinline__ v8f wmma_bf16(v16bf a, v16bf b, v8f c) {
    return __builtin_amdgcn_wmma_f32_16x16x32_bf16(
        /*neg_a=*/false, a, /*neg_b=*/false, b,
        /*c_mod=*/(short)0, c, /*reuse_a=*/false, /*reuse_b=*/false);
}

// A fragment (16x32 bf16): lane covers row M=lane&15; K runs [hf*8,+8) and [16+hf*8,+8)
__device__ __forceinline__ v16bf load_afrag(const __bf16* row, int k0, int hf) {
    v8bf lo = *(const v8bf*)(row + k0 + hf * 8);
    v8bf hi = *(const v8bf*)(row + k0 + 16 + hf * 8);
    return __builtin_shufflevector(lo, hi, 0,1,2,3,4,5,6,7,8,9,10,11,12,13,14,15);
}

// B fragment (32x16 bf16) from W row-major [n][k]: lane covers col N=lane&15;
// K runs [hf*8,+8) and [16+hf*8,+8)
__device__ __forceinline__ v16bf load_bfrag(const __bf16* nrow, int k0, int hf) {
    v8bf lo = *(const v8bf*)(nrow + k0 + hf * 8);
    v8bf hi = *(const v8bf*)(nrow + k0 + 16 + hf * 8);
    return __builtin_shufflevector(lo, hi, 0,1,2,3,4,5,6,7,8,9,10,11,12,13,14,15);
}

// ---- CDNA5 async copy: 16B global -> LDS per lane (ASYNCcnt tracked) --------
__device__ __forceinline__ void async_copy_b128(unsigned lds_addr, const void* gaddr) {
    asm volatile("global_load_async_to_lds_b128 %0, %1, off"
                 :: "v"(lds_addr), "v"(gaddr) : "memory");
}
__device__ __forceinline__ void wait_async0() {
    asm volatile("s_wait_asynccnt 0" ::: "memory");
}

// ---- small elementwise kernels ---------------------------------------------
__global__ void zero_ps_kernel(float* p) {
    if (threadIdx.x < 2 * BB) p[threadIdx.x] = 0.f;
}

__global__ void cvt_bf16_kernel(const float* __restrict__ src, __bf16* __restrict__ dst, int n) {
    int i = blockIdx.x * blockDim.x + threadIdx.x;
    if (i < n) dst[i] = (__bf16)src[i];
}

__global__ void mean4_kernel(const float* __restrict__ x, __bf16* __restrict__ m4) {
    int i = blockIdx.x * 256 + threadIdx.x;
    int row = i >> 10, d = i & 1023;
    const float* p = x + (size_t)row * 4 * DD + d;
    float s = 0.f;
#pragma unroll
    for (int j = 0; j < 4; ++j) s += clipf(p[j * DD], -4.f, 4.f);
    m4[i] = (__bf16)(s * 0.25f);
}

__global__ void mean16_kernel(const __bf16* __restrict__ m4, __bf16* __restrict__ m16) {
    int i = blockIdx.x * 256 + threadIdx.x;
    int row = i >> 10, d = i & 1023;
    float s = 0.f;
#pragma unroll
    for (int j = 0; j < 4; ++j) s += (float)m4[(size_t)(row * 4 + j) * DD + d];
    m16[i] = (__bf16)(s * 0.25f);
}

// ---- fused router: 32 tokens/block, 8 waves x (2 M-tiles x 8 N-tiles) -------
__launch_bounds__(256)
__global__ void router_kernel(const float* __restrict__ x, const float* __restrict__ gum,
                              const __bf16* __restrict__ W1b, const float* __restrict__ b1,
                              const float* __restrict__ W2, const float* __restrict__ b2,
                              float* __restrict__ probsums) {
    __shared__ __align__(16) __bf16 As[32 * DD];   // 64 KB; reused for reductions after K-loop
    float* lsum = (float*)As;        // 64 floats: [32 tokens][2]
    float* psb  = ((float*)As) + 64; // 64 floats: [32 tokens][2]

    const int tok0 = blockIdx.x * 32;
    {   // stage 32 tokens: clip +-4, f32 -> bf16   (128 KB read / block)
        const float4* s4 = (const float4*)(x + (size_t)tok0 * DD);
#pragma unroll
        for (int i = 0; i < 32; ++i) {
            int c = threadIdx.x + i * 256;
            float4 v = s4[c];
            int bofs = c * 4;
            As[bofs + 0] = (__bf16)clipf(v.x, -4.f, 4.f);
            As[bofs + 1] = (__bf16)clipf(v.y, -4.f, 4.f);
            As[bofs + 2] = (__bf16)clipf(v.z, -4.f, 4.f);
            As[bofs + 3] = (__bf16)clipf(v.w, -4.f, 4.f);
        }
    }
    __syncthreads();

    const int lane = threadIdx.x & 31;
    const int wv   = threadIdx.x >> 5;
    const int ml   = lane & 15;
    const int hf   = lane >> 4;
    const int nbase = wv * 128;                    // 8 N-tiles per wave
    const __bf16* arow0 = As + ml * DD;            // M-tile 0: rows 0..15
    const __bf16* arow1 = As + (16 + ml) * DD;     // M-tile 1: rows 16..31

    v8f acc[2][8];
#pragma unroll
    for (int mt = 0; mt < 2; ++mt)
#pragma unroll
        for (int t = 0; t < 8; ++t) acc[mt][t] = (v8f){0.f,0.f,0.f,0.f,0.f,0.f,0.f,0.f};

    for (int k0 = 0; k0 < DD; k0 += 32) {
        v16bf a0 = load_afrag(arow0, k0, hf);
        v16bf a1 = load_afrag(arow1, k0, hf);
#pragma unroll
        for (int t = 0; t < 8; ++t) {
            v16bf b = load_bfrag(W1b + (size_t)(nbase + t * 16 + ml) * DD, k0, hf);
            acc[0][t] = wmma_bf16(a0, b, acc[0][t]);
            acc[1][t] = wmma_bf16(a1, b, acc[1][t]);
        }
    }

    // fused epilogue: relu(h + b1) dotted with the two W2 rows (register-only)
    float p0[2][8], p1[2][8];
#pragma unroll
    for (int mt = 0; mt < 2; ++mt)
#pragma unroll
        for (int r = 0; r < 8; ++r) { p0[mt][r] = 0.f; p1[mt][r] = 0.f; }
#pragma unroll
    for (int t = 0; t < 8; ++t) {
        int n = nbase + t * 16 + ml;
        float w20 = W2[n], w21 = W2[DD + n], bb = b1[n];
#pragma unroll
        for (int mt = 0; mt < 2; ++mt)
#pragma unroll
            for (int r = 0; r < 8; ++r) {
                float h = fmaxf(acc[mt][t][r] + bb, 0.f);
                p0[mt][r] += h * w20;
                p1[mt][r] += h * w21;
            }
    }
#pragma unroll
    for (int mt = 0; mt < 2; ++mt)
#pragma unroll
        for (int r = 0; r < 8; ++r)
#pragma unroll
            for (int mk = 1; mk < 16; mk <<= 1) {
                p0[mt][r] += __shfl_xor(p0[mt][r], mk, 32);
                p1[mt][r] += __shfl_xor(p1[mt][r], mk, 32);
            }

    __syncthreads();                               // all waves done reading As
    if (threadIdx.x < 64) lsum[threadIdx.x] = 0.f; // zero [32][2] logit sums
    __syncthreads();
    if (ml == 0) {
#pragma unroll
        for (int mt = 0; mt < 2; ++mt)
#pragma unroll
            for (int r = 0; r < 8; ++r) {
                int m = mt * 16 + hf * 8 + r;
                atomicAdd(&lsum[2 * m + 0], p0[mt][r]);
                atomicAdd(&lsum[2 * m + 1], p1[mt][r]);
            }
    }
    __syncthreads();

    if (threadIdx.x < 32) {
        int m = threadIdx.x;
        int t = tok0 + m;
        float l0 = clipf(lsum[2 * m + 0] + b2[0], -15.f, 15.f);
        float l1 = clipf(lsum[2 * m + 1] + b2[1], -15.f, 15.f);
        float y0 = (l0 + gum[2 * t + 0]) * 2.0f;   // / TEMP (0.5)
        float y1 = (l1 + gum[2 * t + 1]) * 2.0f;
        float mx = fmaxf(y0, y1);
        float e0 = __expf(y0 - mx), e1 = __expf(y1 - mx);
        float inv = 1.f / (e0 + e1);
        psb[2 * m + 0] = clipf(e0 * inv, 1e-7f, 1.f - 1e-7f);
        psb[2 * m + 1] = clipf(e1 * inv, 1e-7f, 1.f - 1e-7f);
    }
    __syncthreads();
    if (threadIdx.x == 0) {
        float s0 = 0.f, s1 = 0.f;
#pragma unroll
        for (int i = 0; i < 32; ++i) { s0 += psb[2 * i + 0]; s1 += psb[2 * i + 1]; }
        int b = blockIdx.x >> 7;                   // 128 blocks per batch
        atomicAdd(&probsums[2 * b + 0], s0);
        atomicAdd(&probsums[2 * b + 1], s1);
    }
}

// ---- GEMM: out[M,1024] = clip(A W^T + bias, +-6); block = 32 M rows x full N
// 8 waves across N (8 N-tiles each), 2 M-tiles per wave sharing each B fragment
__launch_bounds__(256)
__global__ void gemm_bias_clip_kernel(const __bf16* __restrict__ A, const __bf16* __restrict__ W,
                                      const float* __restrict__ bias, float* __restrict__ out) {
    __shared__ __align__(16) __bf16 Atile[32 * DD];   // 64 KB

    {   // async-stage the 32-row bf16 A block: 4096 x 16B chunks, ASYNCcnt tracked
        const char* gsrc = (const char*)(A + (size_t)blockIdx.x * 32 * DD);
        unsigned lbase = (unsigned)(size_t)&Atile[0];
#pragma unroll
        for (int i = 0; i < 16; ++i) {
            int c = threadIdx.x + i * 256;
            async_copy_b128(lbase + c * 16, gsrc + c * 16);
        }
        wait_async0();
    }
    __syncthreads();

    const int lane = threadIdx.x & 31;
    const int wv   = threadIdx.x >> 5;
    const int ml   = lane & 15;
    const int hf   = lane >> 4;
    const int nbase = wv * 128;
    const __bf16* arow0 = Atile + ml * DD;
    const __bf16* arow1 = Atile + (16 + ml) * DD;

    v8f acc[2][8];
#pragma unroll
    for (int mt = 0; mt < 2; ++mt)
#pragma unroll
        for (int t = 0; t < 8; ++t) acc[mt][t] = (v8f){0.f,0.f,0.f,0.f,0.f,0.f,0.f,0.f};

    for (int k0 = 0; k0 < DD; k0 += 32) {
        v16bf a0 = load_afrag(arow0, k0, hf);
        v16bf a1 = load_afrag(arow1, k0, hf);
#pragma unroll
        for (int t = 0; t < 8; ++t) {
            v16bf b = load_bfrag(W + (size_t)(nbase + t * 16 + ml) * DD, k0, hf);
            acc[0][t] = wmma_bf16(a0, b, acc[0][t]);
            acc[1][t] = wmma_bf16(a1, b, acc[1][t]);
        }
    }

#pragma unroll
    for (int t = 0; t < 8; ++t) {
        int n = nbase + t * 16 + ml;
        float bv = bias[n];
#pragma unroll
        for (int mt = 0; mt < 2; ++mt) {
            int row0 = blockIdx.x * 32 + mt * 16 + hf * 8;
#pragma unroll
            for (int r = 0; r < 8; ++r) {
                out[(size_t)(row0 + r) * DD + n] = clipf(acc[mt][t][r] + bv, -6.f, 6.f);
            }
        }
    }
}

// ---- final mix: out = clip(a*c4 + b*c16_padded, +-6) ------------------------
__global__ void mix_kernel(float* __restrict__ out, const float* __restrict__ c16,
                           const float* __restrict__ probsums) {
    size_t i = (size_t)blockIdx.x * 256 + threadIdx.x;   // 8M elements
    int b   = (int)(i >> 20);
    int row = (int)((i >> 10) & 1023);
    int d   = (int)(i & 1023);
    float w4m  = probsums[2 * b + 0] * (1.0f / LL);
    float w16m = probsums[2 * b + 1] * (1.0f / LL);
    float inv = 1.f / (w4m + w16m + 1e-7f);
    float v = (w4m * inv) * out[i];
    if (row < 256) v += (w16m * inv) * c16[((size_t)b * 256 + row) * DD + d];
    out[i] = clipf(v, -6.f, 6.f);
}

// ---- launcher ---------------------------------------------------------------
extern "C" void kernel_launch(void* const* d_in, const int* in_sizes, int n_in,
                              void* d_out, int out_size, void* d_ws, size_t ws_size,
                              hipStream_t stream) {
    const float* x    = (const float*)d_in[0];
    const float* gum  = (const float*)d_in[1];
    const float* W1   = (const float*)d_in[2];
    const float* b1   = (const float*)d_in[3];
    const float* W2   = (const float*)d_in[4];
    const float* b2   = (const float*)d_in[5];
    const float* W4   = (const float*)d_in[6];
    const float* b4   = (const float*)d_in[7];
    const float* W16  = (const float*)d_in[8];
    const float* b16  = (const float*)d_in[9];
    float* out = (float*)d_out;
    char*  ws  = (char*)d_ws;

    const size_t MB = 1u << 20;
    __bf16* W1b  = (__bf16*)(ws + 0 * MB);    // 2 MB
    __bf16* W4b  = (__bf16*)(ws + 2 * MB);    // 2 MB
    __bf16* W16b = (__bf16*)(ws + 4 * MB);    // 2 MB
    __bf16* m4b  = (__bf16*)(ws + 6 * MB);    // 16 MB (8192 x 1024)
    __bf16* m16b = (__bf16*)(ws + 22 * MB);   // 4 MB  (2048 x 1024)
    float*  c16  = (float*) (ws + 26 * MB);   // 8 MB  (2048 x 1024)
    float*  psum = (float*) (ws + 34 * MB);   // 16 floats

    zero_ps_kernel<<<1, 32, 0, stream>>>(psum);

    cvt_bf16_kernel<<<4096, 256, 0, stream>>>(W1,  W1b,  1 << 20);
    cvt_bf16_kernel<<<4096, 256, 0, stream>>>(W4,  W4b,  1 << 20);
    cvt_bf16_kernel<<<4096, 256, 0, stream>>>(W16, W16b, 1 << 20);

    mean4_kernel <<<32768, 256, 0, stream>>>(x, m4b);
    mean16_kernel<<< 8192, 256, 0, stream>>>(m4b, m16b);

    router_kernel<<<1024, 256, 0, stream>>>(x, gum, W1b, b1, W2, b2, psum);

    gemm_bias_clip_kernel<<<256, 256, 0, stream>>>(m4b,  W4b,  b4,  out);  // M=8192 -> d_out
    gemm_bias_clip_kernel<<< 64, 256, 0, stream>>>(m16b, W16b, b16, c16);  // M=2048 -> ws

    mix_kernel<<<32768, 256, 0, stream>>>(out, c16, psum);
}
